// MultiHeadAttention_23063974379612
// MI455X (gfx1250) — compile-verified, matrix operand order checked by reading, not run on hardware
//
#include <hip/hip_runtime.h>
#include <hip/hip_bf16.h>
#include <stdint.h>

// ---- problem constants (from reference) ----
#define B_  2
#define S_  2048
#define D_  1024
#define H_  16
#define DD_ 64
#define R_  64
#define EPS_ 1e-6f

typedef __attribute__((ext_vector_type(16))) __bf16 v16bf;
typedef __attribute__((ext_vector_type(8)))  __bf16 v8bf;
typedef __attribute__((ext_vector_type(4)))  __bf16 v4bf;
typedef __attribute__((ext_vector_type(8)))  float  v8f;

// -------- CDNA5 async global->LDS copy (guarded; sync fallback) ----------
#if defined(__HIP_DEVICE_COMPILE__) && __has_builtin(__builtin_amdgcn_global_load_async_to_lds_b128)
#define HAVE_ASYNC 1
#else
#define HAVE_ASYNC 0
#endif

__device__ __forceinline__ void copy_b128_g2l(__bf16* lds, const __bf16* g)
{
#if HAVE_ASYNC
    // Builtin signature (from compiler diagnostic): param0 is
    // 'int __attribute__((vector_size(16))) __device__ *'  (v4i addrspace(1)*)
    typedef int v4i_gcc __attribute__((__vector_size__(4 * sizeof(int))));
    typedef __attribute__((address_space(1))) v4i_gcc gv4i;
    typedef __attribute__((address_space(3))) v4i_gcc lv4i;
    __builtin_amdgcn_global_load_async_to_lds_b128(
        (gv4i*)(uintptr_t)g, (lv4i*)(uint32_t)(uintptr_t)lds, 0, 0);
#else
    *(uint4*)lds = *(const uint4*)g;
#endif
}

__device__ __forceinline__ void wait_copies()
{
#if HAVE_ASYNC
#if __has_builtin(__builtin_amdgcn_s_wait_asynccnt)
    __builtin_amdgcn_s_wait_asynccnt(0);
#else
    asm volatile("s_wait_asynccnt 0x0" ::: "memory");
#endif
#endif
}

// =====================================================================
// bf16 GEMM: C[M,N] = (A[M,K] @ B[K,N] + bias)*alpha, B given transposed
// as Bt[N,K].  256 thr = 8 waves; block tile 128x64; wave tile 32x32
// (2x2 WMMA frags -> 4 wmma / k-step, double-buffered LDS).
// =====================================================================
__global__ __launch_bounds__(256)
void gemm_bf16_wmma(const __bf16* __restrict__ A,   // [M,K] row-major
                    const __bf16* __restrict__ Bt,  // [N,K] row-major
                    const float*  __restrict__ bias,
                    float* __restrict__ C,
                    int M, int N, int K, float alpha)
{
    __shared__ __align__(32) __bf16 As[2][128][32];  // 16 KB
    __shared__ __align__(32) __bf16 Bs[2][64][32];   //  8 KB

    const int t    = threadIdx.x;
    const int wave = t >> 5;
    const int lane = t & 31;
    const int m0 = blockIdx.x * 128;
    const int n0 = blockIdx.y * 64;
    const int wm = (wave & 3) * 32;      // wave M offset: 0,32,64,96
    const int wn = (wave >> 2) * 32;     // wave N offset: 0,32
    const int lrow  = lane & 15;
    const int khalf = (lane >> 4) * 16;  // contiguous k-half per lane group (k = 16h + s, same for A and B)

    // staging coordinates (pure b128 copies, bf16 source)
    const int ar = t >> 1;           // 0..127
    const int ac = (t & 1) * 16;     // 0,16
    const int br = t >> 2;           // 0..63
    const int bc = (t & 3) * 8;      // 0,8,16,24

    const __bf16* Abase = A  + (size_t)(m0 + ar) * K + ac;
    const __bf16* Bbase = Bt + (size_t)(n0 + br) * K + bc;

    v8f acc[2][2];
    #pragma unroll
    for (int i = 0; i < 2; ++i)
        #pragma unroll
        for (int j = 0; j < 2; ++j)
            acc[i][j] = (v8f){0.f,0.f,0.f,0.f,0.f,0.f,0.f,0.f};

    auto stage = [&](int buf, int k0) {
        copy_b128_g2l(&As[buf][ar][ac],     Abase + k0);
        copy_b128_g2l(&As[buf][ar][ac + 8], Abase + k0 + 8);
        copy_b128_g2l(&Bs[buf][br][bc],     Bbase + k0);
    };

    stage(0, 0);
    wait_copies();
    __syncthreads();

    const int nk = K >> 5;
    for (int kb = 0; kb < nk; ++kb) {
        const int buf = kb & 1;
        if (kb + 1 < nk) stage(buf ^ 1, (kb + 1) * 32);

        v16bf a0 = *(const v16bf*)&As[buf][wm + lrow][khalf];
        v16bf a1 = *(const v16bf*)&As[buf][wm + 16 + lrow][khalf];
        v16bf b0 = *(const v16bf*)&Bs[buf][wn + lrow][khalf];
        v16bf b1 = *(const v16bf*)&Bs[buf][wn + 16 + lrow][khalf];

        acc[0][0] = __builtin_amdgcn_wmma_f32_16x16x32_bf16(false, a0, false, b0, (short)0, acc[0][0], false, false);
        acc[0][1] = __builtin_amdgcn_wmma_f32_16x16x32_bf16(false, a0, false, b1, (short)0, acc[0][1], false, false);
        acc[1][0] = __builtin_amdgcn_wmma_f32_16x16x32_bf16(false, a1, false, b0, (short)0, acc[1][0], false, false);
        acc[1][1] = __builtin_amdgcn_wmma_f32_16x16x32_bf16(false, a1, false, b1, (short)0, acc[1][1], false, false);

        wait_copies();
        __syncthreads();
    }

    // C layout: lane<16 -> N=lane, VGPR j -> M=j ; lane>=16 -> M=8+j
    #pragma unroll
    for (int mi = 0; mi < 2; ++mi) {
        #pragma unroll
        for (int ni = 0; ni < 2; ++ni) {
            int n  = n0 + wn + ni * 16 + (lane & 15);
            int mb = m0 + wm + mi * 16 + 8 * (lane >> 4);
            float bz = bias[n];
            #pragma unroll
            for (int j = 0; j < 8; ++j)
                C[(size_t)(mb + j) * N + n] = (acc[mi][ni][j] + bz) * alpha;
        }
    }
}

// =====================================================================
// elementwise f32 -> bf16 (8 elems/thread)
// =====================================================================
__global__ __launch_bounds__(256)
void cvt_bf16_kernel(const float* __restrict__ x, __bf16* __restrict__ y, int n8)
{
    int i = blockIdx.x * blockDim.x + threadIdx.x;
    if (i >= n8) return;
    size_t o = (size_t)i * 8;
    float4 f0 = ((const float4*)(x + o))[0];
    float4 f1 = ((const float4*)(x + o))[1];
    v8bf v;
    v[0]=(__bf16)f0.x; v[1]=(__bf16)f0.y; v[2]=(__bf16)f0.z; v[3]=(__bf16)f0.w;
    v[4]=(__bf16)f1.x; v[5]=(__bf16)f1.y; v[6]=(__bf16)f1.z; v[7]=(__bf16)f1.w;
    *(v8bf*)(y + o) = v;
}

// =====================================================================
// weight transpose+convert: W[K,N] f32 -> Wt[N,K] bf16 (32x32 LDS tiles)
// =====================================================================
__global__ __launch_bounds__(256)
void transpose_cvt_kernel(const float* __restrict__ W, __bf16* __restrict__ Wt,
                          int K, int N)
{
    __shared__ float tile[32][33];
    const int t = threadIdx.x;
    const int k0 = blockIdx.x * 32;
    const int n0 = blockIdx.y * 32;
    {
        int kr = t >> 3, nc = (t & 7) * 4;
        float4 f = *(const float4*)(W + (size_t)(k0 + kr) * N + n0 + nc);
        tile[kr][nc+0]=f.x; tile[kr][nc+1]=f.y; tile[kr][nc+2]=f.z; tile[kr][nc+3]=f.w;
    }
    __syncthreads();
    {
        int nr = t >> 3, kc = (t & 7) * 4;
        v4bf v;
        v[0]=(__bf16)tile[kc+0][nr]; v[1]=(__bf16)tile[kc+1][nr];
        v[2]=(__bf16)tile[kc+2][nr]; v[3]=(__bf16)tile[kc+3][nr];
        *(v4bf*)(Wt + (size_t)(n0 + nr) * K + k0 + kc) = v;
    }
}

// =====================================================================
// Row-norm factor: hx[b,h,s] = exp(-0.5 * sum_d Xh[b,s,h*64+d]^2)
// =====================================================================
__global__ __launch_bounds__(256)
void rownorm_kernel(const float* __restrict__ Xh, float* __restrict__ hx)
{
    int i = blockIdx.x * blockDim.x + threadIdx.x;   // over B*H*S
    if (i >= B_ * H_ * S_) return;
    int bh = i / S_;
    int s  = i % S_;
    int b  = bh / H_;
    int h  = bh % H_;
    const float* src = Xh + ((size_t)b * S_ + s) * D_ + h * DD_;
    float acc = 0.f;
    #pragma unroll
    for (int j = 0; j < DD_ / 4; ++j) {
        float4 f = ((const float4*)src)[j];
        acc += f.x*f.x + f.y*f.y + f.z*f.z + f.w*f.w;
    }
    hx[i] = expf(-0.5f * acc);
}

// =====================================================================
// Feature map: phi[bh,s,r] = 0.125 * hx[bh,s] * exp( Xh[b,s,h*64+:] @ wg[h] )
// =====================================================================
__global__ __launch_bounds__(256)
void feature_wmma(const float* __restrict__ Xh, const float* __restrict__ wg,
                  const float* __restrict__ hx, float* __restrict__ phi)
{
    __shared__ __align__(32) __bf16 As[32][32];
    __shared__ __align__(32) __bf16 BsT[64][32];

    const int t    = threadIdx.x;
    const int wave = t >> 5;
    const int lane = t & 31;
    const int bh = blockIdx.y;
    const int b  = bh / H_;
    const int h  = bh % H_;
    const int s0 = blockIdx.x * 32;
    const int tileM = (wave & 1) * 16;
    const int tileN = (wave >> 1) * 16;
    const int lrow  = lane & 15;
    const int khalf = (lane >> 4) * 16;

    const float* wgh = wg + (size_t)h * DD_ * R_;

    v8f acc = {0.f,0.f,0.f,0.f,0.f,0.f,0.f,0.f};

    for (int k0 = 0; k0 < DD_; k0 += 32) {
        {
            int r = t >> 3, c = (t & 7) * 4;
            float4 f = *(const float4*)(Xh + ((size_t)b * S_ + s0 + r) * D_ + h * DD_ + k0 + c);
            __bf16* dst = &As[r][c];
            dst[0]=(__bf16)f.x; dst[1]=(__bf16)f.y; dst[2]=(__bf16)f.z; dst[3]=(__bf16)f.w;
        }
        {
            int kr = t >> 3, nc = (t & 7) * 8;
            const float* src = wgh + (size_t)(k0 + kr) * R_ + nc;
            float4 f0 = ((const float4*)src)[0];
            float4 f1 = ((const float4*)src)[1];
            BsT[nc+0][kr]=(__bf16)f0.x; BsT[nc+1][kr]=(__bf16)f0.y;
            BsT[nc+2][kr]=(__bf16)f0.z; BsT[nc+3][kr]=(__bf16)f0.w;
            BsT[nc+4][kr]=(__bf16)f1.x; BsT[nc+5][kr]=(__bf16)f1.y;
            BsT[nc+6][kr]=(__bf16)f1.z; BsT[nc+7][kr]=(__bf16)f1.w;
        }
        __syncthreads();
        v16bf a  = *(const v16bf*)&As [tileM + lrow][khalf];
        v16bf bb = *(const v16bf*)&BsT[tileN + lrow][khalf];
        acc = __builtin_amdgcn_wmma_f32_16x16x32_bf16(false, a, false, bb,
                                                      (short)0, acc, false, false);
        __syncthreads();
    }
    int n = tileN + (lane & 15);
    int mbase = tileM + 8 * (lane >> 4);
    #pragma unroll
    for (int j = 0; j < 8; ++j) {
        int s = s0 + mbase + j;
        float hv = hx[(size_t)bh * S_ + s];
        phi[((size_t)bh * S_ + s) * R_ + n] = 0.125f * hv * expf(acc[j]);
    }
}

// =====================================================================
// Causal prefix scan; streams kv_prefix (1 GiB) to d_out; attn -> bf16.
// grid = B*H*4 blocks; each owns a 16-wide DD slice of [R=64,DD=64] state.
// =====================================================================
__global__ __launch_bounds__(256)
void scan_kernel(const float* __restrict__ qp,   // [B*H, S, R]
                 const float* __restrict__ kp,   // [B*H, S, R]
                 const float* __restrict__ Vp,   // [B, S, D]
                 float* __restrict__ k_prefix,   // [B*H, S, R]
                 float* __restrict__ kv_prefix,  // [B*H, S, R, DD]
                 __bf16* __restrict__ attn_bf)   // [B, S, D] bf16
{
    const int blk   = blockIdx.x;
    const int bh    = blk >> 2;
    const int slice = blk & 3;
    const int b = bh / H_, h = bh % H_;
    const int d0 = slice * 16;
    const int t = threadIdx.x;

    __shared__ float kps[R_], qps[R_], vs[16], kpfx[R_], qkp[R_];
    __shared__ float red[16][17];
    __shared__ float qksh;

    float st[4] = {0.f, 0.f, 0.f, 0.f};
    if (t < R_) kpfx[t] = 0.f;
    __syncthreads();

    const float* qpb = qp + (size_t)bh * S_ * R_;
    const float* kpb = kp + (size_t)bh * S_ * R_;
    const int dl = t & 15;

    for (int s = 0; s < S_; ++s) {
        if (t < R_) {
            kps[t] = kpb[(size_t)s * R_ + t];
            qps[t] = qpb[(size_t)s * R_ + t];
        } else if (t < R_ + 16) {
            int d = t - R_;
            vs[d] = Vp[((size_t)b * S_ + s) * D_ + h * DD_ + d0 + d];
        }
        __syncthreads();
        if (t < R_) {
            float kc = kpfx[t] + kps[t];
            kpfx[t] = kc;
            if (slice == 0)
                k_prefix[((size_t)bh * S_ + s) * R_ + t] = kc;
            qkp[t] = qps[t] * kc;
        }
        __syncthreads();
        if (t == 0) {
            float q = 0.f;
            #pragma unroll
            for (int i = 0; i < R_; ++i) q += qkp[i];
            qksh = q;
        }
        float pa = 0.f;
        #pragma unroll
        for (int i = 0; i < 4; ++i) {
            int e = t + 256 * i;
            int r = e >> 4;
            float v = st[i] + kps[r] * vs[dl];
            st[i] = v;
            kv_prefix[(((size_t)bh * S_ + s) * R_ + r) * DD_ + d0 + dl] = v;
            pa += qps[r] * v;
        }
        red[t >> 4][dl] = pa;
        __syncthreads();
        if (t < 16) {
            float a = 0.f;
            #pragma unroll
            for (int g = 0; g < 16; ++g) a += red[g][t];
            attn_bf[((size_t)b * S_ + s) * D_ + h * DD_ + d0 + t] =
                (__bf16)(a / (EPS_ + qksh));
        }
        __syncthreads();
    }
}

// =====================================================================
// launch
// =====================================================================
extern "C" void kernel_launch(void* const* d_in, const int* in_sizes, int n_in,
                              void* d_out, int out_size, void* d_ws, size_t ws_size,
                              hipStream_t stream)
{
    const float* v    = (const float*)d_in[0];
    const float* k    = (const float*)d_in[1];
    const float* q    = (const float*)d_in[2];
    const float* wq_w = (const float*)d_in[3];
    const float* wq_b = (const float*)d_in[4];
    const float* wk_w = (const float*)d_in[5];
    const float* wk_b = (const float*)d_in[6];
    const float* wv_w = (const float*)d_in[7];
    const float* wv_b = (const float*)d_in[8];
    const float* wc_w = (const float*)d_in[9];
    const float* wc_b = (const float*)d_in[10];
    const float* wg   = (const float*)d_in[11];

    float* out = (float*)d_out;
    float* k_prefix  = out;                                   // B*H*S*R
    float* kv_prefix = out + (size_t)B_*H_*S_*R_;             // B*H*S*R*DD
    float* out_final = kv_prefix + (size_t)B_*H_*S_*R_*DD_;   // B*S*D

    // workspace carve-up
    const size_t NQ = (size_t)B_ * S_ * D_;        // 4,194,304 elems
    const size_t NP = (size_t)B_ * H_ * S_ * R_;   // 4,194,304
    const size_t NH = (size_t)B_ * H_ * S_;        // 65,536
    const size_t NW = (size_t)D_ * D_;             // 1,048,576

    float* wsf = (float*)d_ws;
    float* Qh   = wsf;  wsf += NQ;
    float* Kh   = wsf;  wsf += NQ;
    float* Vh   = wsf;  wsf += NQ;
    float* qp   = wsf;  wsf += NP;
    float* kpv  = wsf;  wsf += NP;
    float* hq   = wsf;  wsf += NH;
    float* hk   = wsf;  wsf += NH;

    __bf16* wsb = (__bf16*)wsf;
    __bf16* qb      = wsb;  wsb += NQ;
    __bf16* kb      = wsb;  wsb += NQ;
    __bf16* vb      = wsb;  wsb += NQ;
    __bf16* wqT     = wsb;  wsb += NW;
    __bf16* wkT     = wsb;  wsb += NW;
    __bf16* wvT     = wsb;  wsb += NW;
    __bf16* wcT     = wsb;  wsb += NW;
    __bf16* attn_bf = wsb;  wsb += NQ;

    const int M = B_ * S_;                       // 4096
    const float norm_d = 0.35355339059327373f;   // 64^-0.25

    // 0) precision conversion passes (activations + pre-transposed weights)
    int n8 = (int)(NQ / 8);
    cvt_bf16_kernel<<<(n8 + 255)/256, 256, 0, stream>>>(q, qb, n8);
    cvt_bf16_kernel<<<(n8 + 255)/256, 256, 0, stream>>>(k, kb, n8);
    cvt_bf16_kernel<<<(n8 + 255)/256, 256, 0, stream>>>(v, vb, n8);
    dim3 gT(D_/32, D_/32);
    transpose_cvt_kernel<<<gT, 256, 0, stream>>>(wq_w, wqT, D_, D_);
    transpose_cvt_kernel<<<gT, 256, 0, stream>>>(wk_w, wkT, D_, D_);
    transpose_cvt_kernel<<<gT, 256, 0, stream>>>(wv_w, wvT, D_, D_);
    transpose_cvt_kernel<<<gT, 256, 0, stream>>>(wc_w, wcT, D_, D_);

    // 1) Q/K/V projections (bias + scale fused)
    dim3 gGemm(M / 128, D_ / 64);   // 32 x 16
    gemm_bf16_wmma<<<gGemm, 256, 0, stream>>>(qb, wqT, wq_b, Qh, M, D_, D_, norm_d);
    gemm_bf16_wmma<<<gGemm, 256, 0, stream>>>(kb, wkT, wk_b, Kh, M, D_, D_, norm_d);
    gemm_bf16_wmma<<<gGemm, 256, 0, stream>>>(vb, wvT, wv_b, Vh, M, D_, D_, 1.0f);

    // 2) row norms
    int nrows = B_ * H_ * S_;
    rownorm_kernel<<<(nrows + 255) / 256, 256, 0, stream>>>(Qh, hq);
    rownorm_kernel<<<(nrows + 255) / 256, 256, 0, stream>>>(Kh, hk);

    // 3) feature maps
    dim3 gFeat(S_ / 32, B_ * H_);
    feature_wmma<<<gFeat, 256, 0, stream>>>(Qh, wg, hq, qp);
    feature_wmma<<<gFeat, 256, 0, stream>>>(Kh, wg, hk, kpv);

    // 4) causal scan (k_prefix + kv_prefix straight to d_out, attn -> bf16)
    scan_kernel<<<B_ * H_ * 4, 256, 0, stream>>>(qp, kpv, Vh,
                                                 k_prefix, kv_prefix, attn_bf);

    // 5) output projection
    gemm_bf16_wmma<<<gGemm, 256, 0, stream>>>(attn_bf, wcT, wc_b, out_final,
                                              M, D_, D_, 1.0f);
}
